// GNN1_80393197847134
// MI455X (gfx1250) — matrix operand order repaired and checked by reading, not compiled, not run on hardware
//
#include <hip/hip_runtime.h>
#include <math.h>

// ---------------- problem constants (from reference) ----------------
#define N_NODES 50000
#define N_EDGES 800000
#define F       128      // IN_FEATS == H1_FEATS
#define NC      12       // NUM_CLASSES
#define NEG_SLOPE 0.01f
#define ROW_TILES (N_NODES / 16)   // 3125 exactly

typedef __attribute__((ext_vector_type(4)))  _Float16 v4h_t;
typedef __attribute__((ext_vector_type(8)))  _Float16 v8h_t;
typedef __attribute__((ext_vector_type(16))) _Float16 v16h_t;
typedef __attribute__((ext_vector_type(8)))  float    v8f_t;

// ---------------- WMMA fragment loaders (wave32; contiguous-k order used by this code, same for A and B) ----------------
// A (16x32 f16, MxK): lanes 0-15: M=lane, K=k0..k0+15;
//                     lanes 16-31: M=lane-16, K=k0+16..k0+31.
__device__ inline v16h_t load_frag_a(const _Float16* __restrict__ base, int row, int k0, int ksel) {
    const _Float16* p = base + row * F + k0 + 16 * ksel;
    return *(const v16h_t*)(p);
}

// B (32x16 f16, KxN), weights pre-transposed: Wt[n][k] K-contiguous.
// lanes 0-15: N=lane, K=k0..k0+15 ; lanes 16-31: N=lane-16, K=k0+16..k0+31.
__device__ inline v16h_t load_frag_b(const _Float16* __restrict__ wt, int n, int k0, int ksel) {
    return *(const v16h_t*)(wt + n * F + k0 + 16 * ksel);
}

__device__ inline v8f_t wmma_f16(v16h_t a, v16h_t b, v8f_t c) {
    return __builtin_amdgcn_wmma_f32_16x16x32_f16(false, a, false, b, (short)0, c, false, false);
}

// ---------------- prep kernels ----------------
__global__ void cvt_f32_to_f16(const float* __restrict__ in, _Float16* __restrict__ out, int n) {
    for (int i = blockIdx.x * blockDim.x + threadIdx.x; i < n; i += gridDim.x * blockDim.x)
        out[i] = (_Float16)in[i];
}

__global__ void transpose_w_to_f16(const float* __restrict__ W, _Float16* __restrict__ Wt) {
    int i = blockIdx.x * blockDim.x + threadIdx.x;
    if (i < F * F) {
        int k = i / F, n = i % F;
        Wt[n * F + k] = (_Float16)W[k * F + n];
    }
}

__global__ void zero_u32(unsigned* __restrict__ p, int n) {
    for (int i = blockIdx.x * blockDim.x + threadIdx.x; i < n; i += gridDim.x * blockDim.x)
        p[i] = 0u;
}

// ---------------- CSR build: histogram -> scan -> scatter ----------------
__global__ void deg_hist(const int* __restrict__ ei, unsigned* __restrict__ deg) {
    for (int e = blockIdx.x * blockDim.x + threadIdx.x; e < N_EDGES; e += gridDim.x * blockDim.x)
        atomicAdd(&deg[ei[N_EDGES + e]], 1u);
}

__global__ __launch_bounds__(256) void scan_pass1(const unsigned* __restrict__ deg,
                                                  unsigned* __restrict__ starts,
                                                  unsigned* __restrict__ bsum) {
    __shared__ unsigned tmp[256];
    int t = threadIdx.x;
    int i = blockIdx.x * 256 + t;
    unsigned v = (i < N_NODES) ? deg[i] : 0u;
    tmp[t] = v;
    __syncthreads();
#pragma unroll
    for (int off = 1; off < 256; off <<= 1) {
        unsigned add = (t >= off) ? tmp[t - off] : 0u;
        __syncthreads();
        tmp[t] += add;
        __syncthreads();
    }
    if (i < N_NODES) starts[i] = tmp[t] - v;           // exclusive within block
    if (t == 255) bsum[blockIdx.x] = tmp[t];           // block total
}

__global__ __launch_bounds__(256) void scan_pass2(unsigned* __restrict__ bsum, int nb) {
    __shared__ unsigned tmp[256];
    int t = threadIdx.x;
    unsigned v = (t < nb) ? bsum[t] : 0u;
    tmp[t] = v;
    __syncthreads();
#pragma unroll
    for (int off = 1; off < 256; off <<= 1) {
        unsigned add = (t >= off) ? tmp[t - off] : 0u;
        __syncthreads();
        tmp[t] += add;
        __syncthreads();
    }
    if (t < nb) bsum[t] = tmp[t] - v;                  // exclusive
}

__global__ void scan_pass3(unsigned* __restrict__ starts, const unsigned* __restrict__ bsum) {
    int i = blockIdx.x * blockDim.x + threadIdx.x;
    if (i < N_NODES) starts[i] += bsum[i >> 8];
}

__global__ void edge_scatter(const int* __restrict__ ei, const unsigned* __restrict__ starts,
                             unsigned* __restrict__ cursor, int* __restrict__ csr_src) {
    for (int e = blockIdx.x * blockDim.x + threadIdx.x; e < N_EDGES; e += gridDim.x * blockDim.x) {
        int s = ei[e];
        int d = ei[N_EDGES + e];
        unsigned slot = atomicAdd(&cursor[d], 1u);
        csr_src[starts[d] + slot] = s;
    }
}

// ---------------- K1: pooled = relu(x @ W_pool + b_pool), f16 out, SW-pipelined ----------------
__global__ __launch_bounds__(128) void pool_gemm(const _Float16* __restrict__ xh,
                                                 const _Float16* __restrict__ Wpt,
                                                 const float* __restrict__ bias,
                                                 _Float16* __restrict__ pooled) {
    int wave = blockIdx.x * (blockDim.x >> 5) + (threadIdx.x >> 5);
    if (wave >= ROW_TILES) return;            // wave-uniform
    int lane = threadIdx.x & 31;
    int lsub = lane & 15;
    int ksel = lane >> 4;
    int r0 = wave * 16;
    int row = r0 + lsub;

    v8f_t acc[8];
#pragma unroll
    for (int t = 0; t < 8; ++t) { v8f_t z = {}; acc[t] = z; }

    // software pipeline: prefetch (A, B[8]) of step k+1 before WMMAs of step k
    v16h_t a = load_frag_a(xh, row, 0, ksel);
    v16h_t b[8];
#pragma unroll
    for (int t = 0; t < 8; ++t) b[t] = load_frag_b(Wpt, t * 16 + lsub, 0, ksel);

#pragma unroll
    for (int k0 = 0; k0 < F; k0 += 32) {
        v16h_t an;
        v16h_t bn[8];
        if (k0 + 32 < F) {
            an = load_frag_a(xh, row, k0 + 32, ksel);
#pragma unroll
            for (int t = 0; t < 8; ++t) bn[t] = load_frag_b(Wpt, t * 16 + lsub, k0 + 32, ksel);
        }
#pragma unroll
        for (int t = 0; t < 8; ++t) acc[t] = wmma_f16(a, b[t], acc[t]);
        if (k0 + 32 < F) {
            a = an;
#pragma unroll
            for (int t = 0; t < 8; ++t) b[t] = bn[t];
        }
    }

#pragma unroll
    for (int t = 0; t < 8; ++t) {
        int col = t * 16 + lsub;
        float bv = bias[col];
#pragma unroll
        for (int r = 0; r < 8; ++r) {
            float v = acc[t][r] + bv;
            v = v > 0.f ? v : 0.f;
            pooled[(size_t)(r0 + r + 8 * ksel) * F + col] = (_Float16)v;
        }
    }
}

// ---------------- K2: CSR segment-max, one wave per dst, 2-way MLP + prefetch ----------------
__global__ __launch_bounds__(256) void seg_max(const unsigned* __restrict__ starts,
                                               const unsigned* __restrict__ deg,
                                               const int* __restrict__ csr_src,
                                               const _Float16* __restrict__ pooled,
                                               _Float16* __restrict__ hmaxh) {
    int d = blockIdx.x * (blockDim.x >> 5) + (threadIdx.x >> 5);
    if (d >= N_NODES) return;
    int lane = threadIdx.x & 31;
    unsigned s0 = starts[d];
    unsigned n  = deg[d];
    float a0 = 0.f, a1 = 0.f, a2 = 0.f, a3 = 0.f;   // relu output >= 0; deg==0 -> zeros
    unsigned npair = n & ~1u;
    for (unsigned i = 0; i < npair; i += 2) {
        int s1 = csr_src[s0 + i];
        int s2 = csr_src[s0 + i + 1];
        if (i + 2 < npair) {   // prefetch the next pair (gfx1250 global_prefetch_b8)
            int p1 = csr_src[s0 + i + 2];
            int p2 = csr_src[s0 + i + 3];
            __builtin_prefetch(pooled + (size_t)p1 * F + lane * 4, 0, 0);
            __builtin_prefetch(pooled + (size_t)p2 * F + lane * 4, 0, 0);
        }
        v4h_t m1 = *(const v4h_t*)(pooled + (size_t)s1 * F + lane * 4);
        v4h_t m2 = *(const v4h_t*)(pooled + (size_t)s2 * F + lane * 4);
        a0 = fmaxf(fmaxf(a0, (float)m1[0]), (float)m2[0]);
        a1 = fmaxf(fmaxf(a1, (float)m1[1]), (float)m2[1]);
        a2 = fmaxf(fmaxf(a2, (float)m1[2]), (float)m2[2]);
        a3 = fmaxf(fmaxf(a3, (float)m1[3]), (float)m2[3]);
    }
    if (n & 1u) {
        int s = csr_src[s0 + n - 1];
        v4h_t m = *(const v4h_t*)(pooled + (size_t)s * F + lane * 4);
        a0 = fmaxf(a0, (float)m[0]);
        a1 = fmaxf(a1, (float)m[1]);
        a2 = fmaxf(a2, (float)m[2]);
        a3 = fmaxf(a3, (float)m[3]);
    }
    v4h_t o;
    o[0] = (_Float16)a0; o[1] = (_Float16)a1; o[2] = (_Float16)a2; o[3] = (_Float16)a3;
    *(v4h_t*)(hmaxh + (size_t)d * F + lane * 4) = o;
}

// ---------------- K3: fused SAGE + classifier, SW-pipelined over 8 phases ----------------
__device__ inline void load_phase(int p, const _Float16* __restrict__ xh,
                                  const _Float16* __restrict__ hm,
                                  const _Float16* __restrict__ Wst,
                                  const _Float16* __restrict__ Wnt,
                                  int row, int lsub, int ksel,
                                  v16h_t& a, v16h_t (&b)[8]) {
    int k0 = (p >> 1) * 32;
    const _Float16* ab = (p & 1) ? hm : xh;
    const _Float16* wb = (p & 1) ? Wnt : Wst;
    a = load_frag_a(ab, row, k0, ksel);
#pragma unroll
    for (int t = 0; t < 8; ++t) b[t] = load_frag_b(wb, t * 16 + lsub, k0, ksel);
}

__global__ __launch_bounds__(128) void final_fused(const _Float16* __restrict__ xh,
                                                   const _Float16* __restrict__ hmaxh,
                                                   const _Float16* __restrict__ Wst,
                                                   const _Float16* __restrict__ Wnt,
                                                   const float* __restrict__ b_sage,
                                                   const float* __restrict__ W_lin,
                                                   const float* __restrict__ b_lin,
                                                   float* __restrict__ out) {
    __shared__ float hbuf[4][16][F];   // 4 waves * 8KB = 32KB
    int wslot = threadIdx.x >> 5;
    int wave = blockIdx.x * 4 + wslot;
    bool active = (wave < ROW_TILES);  // wave-uniform
    int lane = threadIdx.x & 31;
    int lsub = lane & 15;
    int ksel = lane >> 4;
    int r0 = wave * 16;

    if (active) {
        int row = r0 + lsub;
        v8f_t acc[8];
#pragma unroll
        for (int t = 0; t < 8; ++t) { v8f_t z = {}; acc[t] = z; }

        // 8 phases: (k-step, matrix) pairs; pipeline depth 1
        v16h_t a, b[8];
        load_phase(0, xh, hmaxh, Wst, Wnt, row, lsub, ksel, a, b);
#pragma unroll
        for (int p = 0; p < 8; ++p) {
            v16h_t an;
            v16h_t bn[8];
            if (p < 7) load_phase(p + 1, xh, hmaxh, Wst, Wnt, row, lsub, ksel, an, bn);
#pragma unroll
            for (int t = 0; t < 8; ++t) acc[t] = wmma_f16(a, b[t], acc[t]);
            if (p < 7) {
                a = an;
#pragma unroll
                for (int t = 0; t < 8; ++t) b[t] = bn[t];
            }
        }

#pragma unroll
        for (int t = 0; t < 8; ++t) {
            int col = t * 16 + lsub;
            float bv = b_sage[col];
#pragma unroll
            for (int r = 0; r < 8; ++r) {
                float v = acc[t][r] + bv;
                v = (v >= 0.f) ? v : NEG_SLOPE * v;
                hbuf[wslot][r + 8 * ksel][col] = v;
            }
        }
    }
    __syncthreads();
    if (!active) return;

    // classifier: 16 rows x 12 classes = 192 outputs = 6 per lane
#pragma unroll
    for (int i = 0; i < 6; ++i) {
        int oi = lane + 32 * i;
        int r = oi / NC, c = oi % NC;
        float sum = b_lin[c];
#pragma unroll 4
        for (int k = 0; k < F; ++k)
            sum += hbuf[wslot][r][k] * W_lin[k * NC + c];
        out[(size_t)(r0 + r) * NC + c] = 1.f / (1.f + expf(-sum));
    }
}

// ---------------- launch ----------------
extern "C" void kernel_launch(void* const* d_in, const int* in_sizes, int n_in,
                              void* d_out, int out_size, void* d_ws, size_t ws_size,
                              hipStream_t stream) {
    const float* x      = (const float*)d_in[0];
    const int*   ei     = (const int*)  d_in[1];   // JAX default x64-off: int32
    const float* W_pool = (const float*)d_in[2];
    const float* b_pool = (const float*)d_in[3];
    const float* W_self = (const float*)d_in[4];
    const float* W_neigh= (const float*)d_in[5];
    const float* b_sage = (const float*)d_in[6];
    const float* W_lin  = (const float*)d_in[7];
    const float* b_lin  = (const float*)d_in[8];
    float* out = (float*)d_out;

    // carve workspace (~45 MB)
    char* w = (char*)d_ws;
    auto carve = [&](size_t bytes) { char* p = w; w += (bytes + 255) & ~(size_t)255; return p; };
    _Float16* xh      = (_Float16*)carve((size_t)N_NODES * F * 2);
    _Float16* pooledh = (_Float16*)carve((size_t)N_NODES * F * 2);
    _Float16* hmaxh   = (_Float16*)carve((size_t)N_NODES * F * 2);
    _Float16* Wpt     = (_Float16*)carve((size_t)F * F * 2);
    _Float16* Wst     = (_Float16*)carve((size_t)F * F * 2);
    _Float16* Wnt     = (_Float16*)carve((size_t)F * F * 2);
    unsigned* deg     = (unsigned*)carve((size_t)N_NODES * 4);
    unsigned* starts  = (unsigned*)carve((size_t)N_NODES * 4);
    unsigned* cursor  = (unsigned*)carve((size_t)N_NODES * 4);
    unsigned* bsum    = (unsigned*)carve(256 * 4);
    int*      csr_src = (int*)     carve((size_t)N_EDGES * 4);

    const int NF = N_NODES * F;
    const int NB = (N_NODES + 255) / 256;   // 196 scan blocks

    // precision prep
    cvt_f32_to_f16<<<2048, 256, 0, stream>>>(x, xh, NF);
    transpose_w_to_f16<<<(F * F + 255) / 256, 256, 0, stream>>>(W_pool, Wpt);
    transpose_w_to_f16<<<(F * F + 255) / 256, 256, 0, stream>>>(W_self, Wst);
    transpose_w_to_f16<<<(F * F + 255) / 256, 256, 0, stream>>>(W_neigh, Wnt);

    // CSR build (atomic-light)
    zero_u32<<<NB, 256, 0, stream>>>(deg, N_NODES);
    zero_u32<<<NB, 256, 0, stream>>>(cursor, N_NODES);
    deg_hist<<<1024, 256, 0, stream>>>(ei, deg);
    scan_pass1<<<NB, 256, 0, stream>>>(deg, starts, bsum);
    scan_pass2<<<1, 256, 0, stream>>>(bsum, NB);
    scan_pass3<<<NB, 256, 0, stream>>>(starts, bsum);
    edge_scatter<<<1024, 256, 0, stream>>>(ei, starts, cursor, csr_src);

    // compute
    pool_gemm<<<(ROW_TILES + 3) / 4, 128, 0, stream>>>(xh, Wpt, b_pool, pooledh);
    seg_max<<<(N_NODES + 7) / 8, 256, 0, stream>>>(starts, deg, csr_src, pooledh, hmaxh);
    final_fused<<<(ROW_TILES + 3) / 4, 128, 0, stream>>>(xh, hmaxh, Wst, Wnt,
                                                         b_sage, W_lin, b_lin, out);
}
